// MultiHeadBlock_11484742549736
// MI455X (gfx1250) — matrix operand order repaired and checked by reading, not compiled, not run on hardware
//
#include <hip/hip_runtime.h>
#include <hip/hip_bf16.h>
#include <math.h>
#include <stdint.h>

// ---------------------------------------------------------------------------
// MultiHeadBlock for MI455X (gfx1250): bf16 WMMA everywhere, f32 accumulate.
//  B=2, S=2048, D=1024, H=16, DK=64
// Pipeline (all on `stream`, scratch reused sequentially):
//   1) gemm(q,Wq)+bq -> buf0(f32)      4) gemm(v,Wv)+bv -> buf0 ; packV -> Vt
//   2) ropePack(buf0) -> Qb(bf16)      5) flashAttn(Qb,Kb,Vt) -> buf0(f32)
//   3) gemm(k,Wk)+bk -> buf0 ; ropePack -> Kb
//   6) gemm(buf0,Wo)+bo -> d_out
// ---------------------------------------------------------------------------

typedef __attribute__((ext_vector_type(16))) __bf16 v16bf;
typedef __attribute__((ext_vector_type(2)))  __bf16 bf16x2;
typedef __attribute__((ext_vector_type(8)))  float  v8f;
typedef __attribute__((ext_vector_type(4)))  int    v4i;

constexpr int Bc  = 2;
constexpr int Sc  = 2048;
constexpr int Dc  = 1024;
constexpr int Hc  = 16;
constexpr int DKc = 64;
constexpr int Mrows = Bc * Sc;   // 4096

// ---- CDNA5 async global->LDS (guarded; falls back to manual staging) ------
#if defined(__has_builtin)
#if __has_builtin(__builtin_amdgcn_global_load_async_to_lds_b128)
#define USE_ASYNC_LDS 1
#endif
#endif

#ifdef USE_ASYNC_LDS
typedef __attribute__((address_space(1))) v4i gas_v4i;   // global int4
typedef __attribute__((address_space(3))) v4i las_v4i;   // LDS int4
__device__ __forceinline__ void async_copy16(const void* g, void* l) {
    // generic->AS1 keeps the 64-bit VA; generic LDS VA low 32 bits == LDS offset.
    __builtin_amdgcn_global_load_async_to_lds_b128(
        (gas_v4i*)(uintptr_t)g, (las_v4i*)(uint32_t)(uintptr_t)l, 0, 0);
}
#endif

__device__ __forceinline__ void wait_async0() {
#if defined(__has_builtin) && __has_builtin(__builtin_amdgcn_s_wait_asynccnt)
    __builtin_amdgcn_s_wait_asynccnt(0);
#else
    asm volatile("s_wait_asynccnt 0x0" ::: "memory");
#endif
}

// ------------------------------ GEMM --------------------------------------
// Y[4096,1024] = X[4096,1024] @ W[1024,1024] + bias ; bf16 WMMA compute.
// Block tile 128x128, 8 waves (2 along M x 4 along N), wave tile 64x32:
// 8 WMMAs per K-step per wave.  Dims are compile-time so stores use
// immediate offsets instead of 64-bit address chains.
constexpr int BM = 128, BN = 128, BK = 32;
constexpr int LDSA = BK + 2;   // 34 elems (68 B): 17-dword stride, conflict-free
constexpr int LDSB = BK + 2;

__global__ void __launch_bounds__(256)
gemm_bf16_wmma(const float* __restrict__ X, const float* __restrict__ W,
               const float* __restrict__ bias, float* __restrict__ Y)
{
    __shared__ __bf16 As[BM * LDSA];       // [m][k]
    __shared__ __bf16 Bs[BN * LDSB];       // [n][k]  (W transposed on load)

    const int tid  = threadIdx.x;
    const int lane = tid & 31;
    const int wave = tid >> 5;             // 8 waves
    const int wm   = wave & 1;             // 2 waves along M (64 rows each)
    const int wn   = wave >> 1;            // 4 waves along N (32 cols each)
    const int m0   = blockIdx.y * BM;
    const int n0   = blockIdx.x * BN;
    const int half = lane >> 4;
    const int lr   = lane & 15;

    v8f acc[4][2] = {};

    for (int k0 = 0; k0 < Dc; k0 += BK) {
        __syncthreads();
        // ---- stage A tile: 128x32 f32 -> bf16 (float4 coalesced) ----
        #pragma unroll
        for (int i = 0; i < 4; ++i) {
            int idx = tid + i * 256;               // 1024 float4
            int row = idx >> 3;
            int kg  = (idx & 7) << 2;
            const float* src = X + (size_t)(m0 + row) * Dc + k0 + kg;
            const float4 f = *(const float4*)src;
            if (k0 + BK < Dc) __builtin_prefetch(src + BK, 0, 1);
            bf16x2 p01 = { (__bf16)f.x, (__bf16)f.y };
            bf16x2 p23 = { (__bf16)f.z, (__bf16)f.w };
            *(bf16x2*)&As[row * LDSA + kg]     = p01;
            *(bf16x2*)&As[row * LDSA + kg + 2] = p23;
        }
        // ---- stage B tile transposed: W[k0..+31][n0..+127] -> Bs[n][k] ----
        #pragma unroll
        for (int i = 0; i < 4; ++i) {
            int idx = tid + i * 256;               // 1024 float4
            int kk  = idx >> 5;
            int ng  = (idx & 31) << 2;
            const float4 f = *(const float4*)(W + (size_t)(k0 + kk) * Dc + n0 + ng);
            Bs[(ng + 0) * LDSB + kk] = (__bf16)f.x;
            Bs[(ng + 1) * LDSB + kk] = (__bf16)f.y;
            Bs[(ng + 2) * LDSB + kk] = (__bf16)f.z;
            Bs[(ng + 3) * LDSB + kk] = (__bf16)f.w;
        }
        __syncthreads();

        // ---- fragments (ISA 16-bit A 16x32 / B 32x16 layouts) ----
        v16bf a[4], b[2];
        #pragma unroll
        for (int tm = 0; tm < 4; ++tm) {
            int m = wm * 64 + tm * 16 + lr;
            #pragma unroll
            for (int v = 0; v < 8; ++v) {
                int K = ((v >> 2) << 4) + (half << 3) + ((v & 3) << 1);
                bf16x2 p = *(const bf16x2*)&As[m * LDSA + K];
                a[tm][2 * v] = p.x; a[tm][2 * v + 1] = p.y;
            }
        }
        #pragma unroll
        for (int tn = 0; tn < 2; ++tn) {
            int n = wn * 32 + tn * 16 + lr;
            #pragma unroll
            for (int v = 0; v < 8; ++v) {
                int K = ((v >> 2) << 4) + (half << 3) + ((v & 3) << 1);
                bf16x2 p = *(const bf16x2*)&Bs[n * LDSB + K];
                b[tn][2 * v] = p.x; b[tn][2 * v + 1] = p.y;
            }
        }
        #pragma unroll
        for (int tm = 0; tm < 4; ++tm)
            #pragma unroll
            for (int tn = 0; tn < 2; ++tn)
                acc[tm][tn] = __builtin_amdgcn_wmma_f32_16x16x32_bf16(
                    false, a[tm], false, b[tn], (short)0, acc[tm][tn], false, false);
    }

    // ---- epilogue: C layout M = r + 8*half, N = lane&15 ----
    #pragma unroll
    for (int tm = 0; tm < 4; ++tm)
        #pragma unroll
        for (int tn = 0; tn < 2; ++tn) {
            int n = n0 + wn * 32 + tn * 16 + lr;
            int m = m0 + wm * 64 + tm * 16 + 8 * half;
            float bv = bias[n];
            float* yp = Y + (size_t)m * Dc + n;
            #pragma unroll
            for (int r = 0; r < 8; ++r)
                yp[(size_t)r * Dc] = acc[tm][tn][r] + bv;   // immediate offsets
        }
}

// --------------------------- RoPE + pack -----------------------------------
// P f32 [M, D] -> Out bf16 [B, H, S, DK] with rotate-half RoPE.
__global__ void __launch_bounds__(256)
rope_pack(const float* __restrict__ P, __bf16* __restrict__ Out)
{
    int idx = blockIdx.x * blockDim.x + threadIdx.x;   // over M*D
    int col = idx & (Dc - 1);
    int row = idx / Dc;
    int b = row / Sc, s = row & (Sc - 1);
    int h = col / DKc, d = col & (DKc - 1);
    int j = d & 31;
    float inv = __expf(-((float)(2 * j) / 64.f) * __logf(10000.f));
    float ang = (float)s * inv;
    float c, sn;
    __sincosf(ang, &sn, &c);
    float x = P[idx];
    float other = (d < 32) ? -P[(size_t)row * Dc + col + 32]
                           :  P[(size_t)row * Dc + col - 32];
    float y = x * c + other * sn;
    Out[(((size_t)(b * Hc + h) * Sc + s) * DKc) + d] = (__bf16)y;
}

// V f32 [M, D] -> Vt bf16 [B, H, DK, S] (transposed for PV B-fragments).
__global__ void __launch_bounds__(256)
pack_v(const float* __restrict__ P, __bf16* __restrict__ Vt)
{
    int idx = blockIdx.x * blockDim.x + threadIdx.x;
    int col = idx & (Dc - 1);
    int row = idx / Dc;
    int b = row / Sc, s = row & (Sc - 1);
    int h = col / DKc, d = col & (DKc - 1);
    Vt[(((size_t)(b * Hc + h) * DKc + d) * Sc) + s] = (__bf16)P[idx];
}

// --------------------------- Flash attention -------------------------------
// grid (S/64, H, B), block 128 (4 waves x 16 q rows). Causal, online softmax.
// K/V chunks staged with CDNA5 async global->LDS DMA when available.
constexpr int QT  = 64;
constexpr int KVC = 32;
constexpr int LDK = DKc + 8;   // 72 elems = 144 B: 16B-aligned rows, no conflicts
constexpr int LDV = KVC + 8;   // 40 elems = 80 B:  16B-aligned rows, no conflicts
constexpr int LDP = KVC + 2;   // 34

__global__ void __launch_bounds__(128)
flash_attn(const __bf16* __restrict__ Qb, const __bf16* __restrict__ Kb,
           const __bf16* __restrict__ Vt, float* __restrict__ O)
{
    __shared__ __bf16 Ks[KVC * LDK];        // [kv][dk]
    __shared__ __bf16 Vs[DKc * LDV];        // [dk][kv]
    __shared__ __bf16 Ps[4 * 16 * LDP];     // per-wave P tile [16][32]

    const int tid  = threadIdx.x;
    const int lane = tid & 31;
    const int wave = tid >> 5;
    const int half = lane >> 4;
    const int lr   = lane & 15;
    const int qt = blockIdx.x, h = blockIdx.y, b = blockIdx.z;
    const size_t bh = (size_t)(b * Hc + h);
    const __bf16* Qp = Qb + bh * Sc * DKc;
    const __bf16* Kp = Kb + bh * Sc * DKc;
    const __bf16* Vp = Vt + bh * DKc * Sc;
    const int q0 = qt * QT + wave * 16;

    // Q A-fragments for both 32-wide halves of DK=64
    v16bf aq[2];
    #pragma unroll
    for (int kc = 0; kc < 2; ++kc)
        #pragma unroll
        for (int v = 0; v < 8; ++v) {
            int K = kc * 32 + ((v >> 2) << 4) + (half << 3) + ((v & 3) << 1);
            bf16x2 p = *(const bf16x2*)(Qp + (size_t)(q0 + lr) * DKc + K);
            aq[kc][2 * v] = p.x; aq[kc][2 * v + 1] = p.y;
        }

    v8f accO[4] = {};
    float mrow[8], lsum[8];
    #pragma unroll
    for (int r = 0; r < 8; ++r) { mrow[r] = -3.0e38f; lsum[r] = 0.f; }

    const int kv_end = qt * QT + QT;   // causal bound for this block
    for (int kv0 = 0; kv0 < kv_end; kv0 += KVC) {
        __syncthreads();
#ifdef USE_ASYNC_LDS
        // ---- async DMA stage: K chunk 32x64 (4 KB), V chunk 64x32 (4 KB) ----
        #pragma unroll
        for (int i = 0; i < 2; ++i) {
            int idx = tid + i * 128;            // 256 x 16B chunks
            int kv = idx >> 3;
            int c  = (idx & 7) << 3;            // 8 bf16 per chunk
            async_copy16(Kp + (size_t)(kv0 + kv) * DKc + c, &Ks[kv * LDK + c]);
        }
        #pragma unroll
        for (int i = 0; i < 2; ++i) {
            int idx = tid + i * 128;
            int d = idx >> 2;
            int c = (idx & 3) << 3;
            async_copy16(Vp + (size_t)d * Sc + kv0 + c, &Vs[d * LDV + c]);
        }
        wait_async0();
#else
        // ---- fallback: through-register staging ----
        #pragma unroll
        for (int i = 0; i < 8; ++i) {
            int idx = tid + i * 128;
            int kv = idx >> 5;
            int dg = (idx & 31) << 1;
            bf16x2 p = *(const bf16x2*)(Kp + (size_t)(kv0 + kv) * DKc + dg);
            *(bf16x2*)&Ks[kv * LDK + dg] = p;
        }
        #pragma unroll
        for (int i = 0; i < 8; ++i) {
            int idx = tid + i * 128;
            int d  = idx >> 4;
            int kg = (idx & 15) << 1;
            bf16x2 p = *(const bf16x2*)(Vp + (size_t)d * Sc + kv0 + kg);
            *(bf16x2*)&Vs[d * LDV + kg] = p;
        }
#endif
        __syncthreads();

        // ---- scores: 16q x 32kv as two 16x16 WMMA tiles, K-dim 64 ----
        v8f sc[2];
        #pragma unroll
        for (int tn = 0; tn < 2; ++tn) {
            v8f c = {};
            #pragma unroll
            for (int kc = 0; kc < 2; ++kc) {
                v16bf bk;
                #pragma unroll
                for (int v = 0; v < 8; ++v) {
                    int K = kc * 32 + ((v >> 2) << 4) + (half << 3) + ((v & 3) << 1);
                    bf16x2 p = *(const bf16x2*)&Ks[(tn * 16 + lr) * LDK + K];
                    bk[2 * v] = p.x; bk[2 * v + 1] = p.y;
                }
                c = __builtin_amdgcn_wmma_f32_16x16x32_bf16(
                        false, aq[kc], false, bk, (short)0, c, false, false);
            }
            sc[tn] = c;
        }

        // ---- causal mask + online softmax (row = r + 8*half) ----
        #pragma unroll
        for (int r = 0; r < 8; ++r) {
            int qrow = q0 + r + 8 * half;
            float mx = -3.0e38f;
            #pragma unroll
            for (int tn = 0; tn < 2; ++tn) {
                int kvp = kv0 + tn * 16 + lr;
                float s_ = sc[tn][r] * 0.125f;              // 1/sqrt(64)
                if (kvp > qrow) s_ = -1.0e9f;
                sc[tn][r] = s_;
                mx = fmaxf(mx, s_);
            }
            #pragma unroll
            for (int off = 1; off < 16; off <<= 1)
                mx = fmaxf(mx, __shfl_xor(mx, off, 32));
            float mnew = fmaxf(mrow[r], mx);
            float sum = 0.f;
            #pragma unroll
            for (int tn = 0; tn < 2; ++tn) {
                float p = __expf(sc[tn][r] - mnew);
                sc[tn][r] = p;
                sum += p;
            }
            #pragma unroll
            for (int off = 1; off < 16; off <<= 1)
                sum += __shfl_xor(sum, off, 32);
            float corr = __expf(mrow[r] - mnew);
            lsum[r] = lsum[r] * corr + sum;
            mrow[r] = mnew;
            #pragma unroll
            for (int t = 0; t < 4; ++t) accO[t][r] *= corr;
        }

        // ---- P: C layout -> LDS -> A layout (same-wave in-order LDS) ----
        __bf16* Pw = &Ps[wave * 16 * LDP];
        #pragma unroll
        for (int tn = 0; tn < 2; ++tn)
            #pragma unroll
            for (int r = 0; r < 8; ++r)
                Pw[(r + 8 * half) * LDP + tn * 16 + lr] = (__bf16)sc[tn][r];
        asm volatile("s_wait_dscnt 0" ::: "memory");
        v16bf pa;
        #pragma unroll
        for (int v = 0; v < 8; ++v) {
            int K = ((v >> 2) << 4) + (half << 3) + ((v & 3) << 1);
            bf16x2 p = *(const bf16x2*)&Pw[lr * LDP + K];
            pa[2 * v] = p.x; pa[2 * v + 1] = p.y;
        }

        // ---- accO += P @ V (4 N-tiles over DK=64) ----
        #pragma unroll
        for (int t = 0; t < 4; ++t) {
            v16bf bv;
            #pragma unroll
            for (int v = 0; v < 8; ++v) {
                int K = ((v >> 2) << 4) + (half << 3) + ((v & 3) << 1);
                bf16x2 p = *(const bf16x2*)&Vs[(t * 16 + lr) * LDV + K];
                bv[2 * v] = p.x; bv[2 * v + 1] = p.y;
            }
            accO[t] = __builtin_amdgcn_wmma_f32_16x16x32_bf16(
                          false, pa, false, bv, (short)0, accO[t], false, false);
        }
    }

    // ---- epilogue: normalize, scatter to [b*S+s][h*64+d] f32 ----
    #pragma unroll
    for (int t = 0; t < 4; ++t)
        #pragma unroll
        for (int r = 0; r < 8; ++r) {
            int s_ = q0 + r + 8 * half;
            int d  = t * 16 + lr;
            O[((size_t)(b * Sc + s_)) * Dc + h * DKc + d] =
                accO[t][r] * (1.f / lsum[r]);
        }
}

// ------------------------------ launch -------------------------------------
extern "C" void kernel_launch(void* const* d_in, const int* in_sizes, int n_in,
                              void* d_out, int out_size, void* d_ws, size_t ws_size,
                              hipStream_t stream)
{
    const float* q   = (const float*)d_in[0];
    const float* k   = (const float*)d_in[1];
    const float* v   = (const float*)d_in[2];
    // d_in[3] = mask (causal handled analytically)
    const float* Wq  = (const float*)d_in[4];
    const float* bq  = (const float*)d_in[5];
    const float* Wk  = (const float*)d_in[6];
    const float* bk  = (const float*)d_in[7];
    const float* Wv  = (const float*)d_in[8];
    const float* bv  = (const float*)d_in[9];
    const float* Wo  = (const float*)d_in[10];
    const float* bo  = (const float*)d_in[11];
    float* out = (float*)d_out;

    // workspace layout (40 MB total)
    char* ws = (char*)d_ws;
    float*  buf0 = (float*)ws;                                  // 16 MB f32 scratch
    __bf16* Qb   = (__bf16*)(ws + (size_t)16 * 1024 * 1024);    // 8 MB
    __bf16* Kbuf = (__bf16*)(ws + (size_t)24 * 1024 * 1024);    // 8 MB
    __bf16* Vt   = (__bf16*)(ws + (size_t)32 * 1024 * 1024);    // 8 MB

    dim3 gGemm(Dc / BN, Mrows / BM);            // (8, 32)
    dim3 gElem((Mrows * Dc) / 256);             // 16384 blocks
    dim3 gAttn(Sc / QT, Hc, Bc);                // (32, 16, 2)

    // Q projection + RoPE pack
    gemm_bf16_wmma<<<gGemm, 256, 0, stream>>>(q, Wq, bq, buf0);
    rope_pack<<<gElem, 256, 0, stream>>>(buf0, Qb);
    // K projection + RoPE pack
    gemm_bf16_wmma<<<gGemm, 256, 0, stream>>>(k, Wk, bk, buf0);
    rope_pack<<<gElem, 256, 0, stream>>>(buf0, Kbuf);
    // V projection + transpose pack
    gemm_bf16_wmma<<<gGemm, 256, 0, stream>>>(v, Wv, bv, buf0);
    pack_v<<<gElem, 256, 0, stream>>>(buf0, Vt);
    // Flash attention -> buf0 as [M, D] f32
    flash_attn<<<gAttn, 128, 0, stream>>>(Qb, Kbuf, Vt, buf0);
    // Output projection
    gemm_bf16_wmma<<<gGemm, 256, 0, stream>>>(buf0, Wo, bo, out);
}
